// GCN_65524021068099
// MI455X (gfx1250) — matrix operand order repaired and checked by reading, not compiled, not run on hardware
//
#include <hip/hip_runtime.h>

// ---------------------------------------------------------------------------
// GCN forward for MI455X (gfx1250, wave32).
//  - dense transforms: v_wmma_f32_16x16x32_bf16, B fragments pinned in VGPRs
//  - GEMM1 A tiles: pre-converted bf16, staged via async global->LDS copies
//    (global_load_async_to_lds_b128 / s_wait_asynccnt), double-buffered to
//    overlap the copy of tile t+1 with the WMMA work on tile t
//  - software-pipelined fragment loads (chunk c+1 issued before WMMAs of c)
//  - sparse normalized aggregation: wave-per-edge global_atomic_add_f32
// ---------------------------------------------------------------------------

typedef __attribute__((ext_vector_type(16))) __bf16 v16bf;
typedef __attribute__((ext_vector_type(8)))  float  v8f;

#define NN  100000   // nodes
#define NE  1600000  // edges (excluding self loops)
#define CIN 512
#define CH  128
#define CL  40

// ---- bf16 helpers ---------------------------------------------------------

__device__ __forceinline__ unsigned int pack_bf16(float lo, float hi) {
#if __has_builtin(__builtin_amdgcn_cvt_pk_bf16_f32)
  typedef __attribute__((ext_vector_type(2))) __bf16 v2bf;
  union { v2bf v; unsigned int u; } r;
  r.v = __builtin_amdgcn_cvt_pk_bf16_f32(lo, hi);
  return r.u;
#else
  union { __bf16 b[2]; unsigned int u; } r;   // lowers to v_cvt_pk_bf16_f32
  r.b[0] = (__bf16)lo;
  r.b[1] = (__bf16)hi;
  return r.u;
#endif
}

union Frag16 {
  v16bf          v;
  unsigned short h[16];
  unsigned int   u[8];
};

// ---- CDNA5 async global->LDS copy (ASYNCcnt) ------------------------------

__device__ __forceinline__ void wait_async0() {
#if __has_builtin(__builtin_amdgcn_s_wait_asynccnt)
  __builtin_amdgcn_s_wait_asynccnt(0);
#else
  asm volatile("s_wait_asynccnt 0x0" ::: "memory");
#endif
}

// copy 128 contiguous bytes (64 bf16) global -> LDS, asynchronously.
// ISA 15.18.3: VDST = LDS byte offset, VADDR = 64-bit global address, "off" =
// no SADDR; the immediate offset is added to BOTH the LDS and global address.
__device__ __forceinline__ void stage128B_async(const unsigned short* g,
                                                const unsigned short* l) {
  const unsigned ldsoff = (unsigned)(unsigned long long)l;  // low 32 = LDS off
#pragma unroll
  for (int q = 0; q < 8; ++q)
    asm volatile("global_load_async_to_lds_b128 %0, %1, off offset:%2"
                 :: "v"(ldsoff), "v"(g), "i"(q * 16) : "memory");
}

// --------------------------- degree / norm ---------------------------------

__global__ void k_deg_init(float* __restrict__ deg) {
  int i = blockIdx.x * blockDim.x + threadIdx.x;
  if (i < NN) deg[i] = 1.0f;                 // self loop
}

__global__ void k_deg_edges(const long long* __restrict__ dst,
                            float* __restrict__ deg) {
  int e = blockIdx.x * blockDim.x + threadIdx.x;
  if (e < NE) unsafeAtomicAdd(&deg[(int)dst[e]], 1.0f);
}

__global__ void k_dinv(const float* __restrict__ deg, float* __restrict__ dinv) {
  int i = blockIdx.x * blockDim.x + threadIdx.x;
  if (i < NN) {
    float d = deg[i];
    dinv[i] = (d > 0.0f) ? rsqrtf(d) : 0.0f;
  }
}

// ------------------- streaming fp32 -> bf16 convert of x -------------------

__global__ void k_cvt_x(const float* __restrict__ x,
                        uint4* __restrict__ xbf4) {
  int p = blockIdx.x * blockDim.x + threadIdx.x;
  if (p < NN * CIN / 8) {
    const float4* s = (const float4*)x + (size_t)p * 2;
    float4 f = s[0], g = s[1];
    xbf4[p] = make_uint4(pack_bf16(f.x, f.y), pack_bf16(f.z, f.w),
                         pack_bf16(g.x, g.y), pack_bf16(g.z, g.w));
  }
}

// --------------------------- GEMM1: hlin = x @ W1 --------------------------

__device__ __forceinline__ void load_a_pair(const unsigned short* cur, int c,
                                            int mr, int half,
                                            Frag16& f0, Frag16& f1) {
  const int kb0 = mr * CIN + 32 * c + 8 * half;
  const int kb1 = (16 + mr) * CIN + 32 * c + 8 * half;
  *(uint4*)&f0.u[0] = *(const uint4*)&cur[kb0];        // ds_load_b128
  *(uint4*)&f0.u[4] = *(const uint4*)&cur[kb0 + 16];
  *(uint4*)&f1.u[0] = *(const uint4*)&cur[kb1];
  *(uint4*)&f1.u[4] = *(const uint4*)&cur[kb1 + 16];
}

__global__ __launch_bounds__(256) void k_gemm1(const unsigned short* __restrict__ xbf,
                                               const float* __restrict__ W1,
                                               float* __restrict__ hlin) {
  __shared__ unsigned short lds_a[2][32 * CIN];   // 2 x 32 KB
  const int lane = threadIdx.x & 31;
  const int wave = threadIdx.x >> 5;              // N-tile 0..7
  const int half = lane >> 4;
  const int mr   = lane & 15;

  // B fragments: 16 K-chunks of 32; element i<8 holds K = 8*half+i, i>=8 holds K = 16+8*half+(i-8).
  Frag16 bfr[16];
  {
    const int n = wave * 16 + mr;
#pragma unroll
    for (int c = 0; c < 16; ++c)
#pragma unroll
      for (int j = 0; j < 8; ++j) {
        const int k = 32 * c + 8 * half + 2 * j + (j >= 4 ? 8 : 0);
        bfr[c].u[j] = pack_bf16(W1[k * CH + n], W1[(k + 1) * CH + n]);
      }
  }

  const int tiles = NN / 32;                      // 3125, exact
  const int slice = threadIdx.x * 64;             // 64 bf16 = 128 B per thread

  // prologue: stage first tile into buffer 0
  stage128B_async(xbf + (size_t)blockIdx.x * 32 * CIN + slice, &lds_a[0][slice]);

  int buf = 0;
  for (int tile = blockIdx.x; tile < tiles; tile += gridDim.x) {
    wait_async0();                                // this wave's copies landed
    __syncthreads();                              // everyone's copies landed

    const int next = tile + gridDim.x;            // prefetch next tile
    if (next < tiles)
      stage128B_async(xbf + (size_t)next * 32 * CIN + slice,
                      &lds_a[buf ^ 1][slice]);

    const unsigned short* cur = lds_a[buf];
    v8f acc0 = {}, acc1 = {};
    Frag16 a0, a1;
    load_a_pair(cur, 0, mr, half, a0, a1);        // preload chunk 0
#pragma unroll
    for (int c = 0; c < 16; ++c) {
      Frag16 n0, n1;                              // software pipeline: issue
      if (c < 15)                                 // chunk c+1 loads first
        load_a_pair(cur, c + 1, mr, half, n0, n1);
      acc0 = __builtin_amdgcn_wmma_f32_16x16x32_bf16(false, a0.v, false, bfr[c].v,
                                                     (short)0, acc0, false, false);
      acc1 = __builtin_amdgcn_wmma_f32_16x16x32_bf16(false, a1.v, false, bfr[c].v,
                                                     (short)0, acc1, false, false);
      if (c < 15) { a0 = n0; a1 = n1; }
    }
    const int m0 = tile * 32;
#pragma unroll
    for (int r = 0; r < 8; ++r) {
      const int row0 = m0 + r + 8 * half;          // C layout: VGPR r -> M=r(+8)
      const int row1 = m0 + 16 + r + 8 * half;
      hlin[(size_t)row0 * CH + wave * 16 + mr] = acc0[r];
      hlin[(size_t)row1 * CH + wave * 16 + mr] = acc1[r];
    }
    buf ^= 1;
  }
}

// ------------------------- layer-1 aggregation -----------------------------

__global__ void k_selfinit1(const float* __restrict__ hlin,
                            const float* __restrict__ dinv,
                            float* __restrict__ agg) {
  int idx = blockIdx.x * blockDim.x + threadIdx.x;
  if (idx < NN * CH) {
    int i = idx >> 7;
    float dv = dinv[i];
    agg[idx] = hlin[idx] * dv * dv;            // self-loop term
  }
}

// one wave per edge: 32 lanes x float4 = 128 channels
__global__ void k_agg1(const long long* __restrict__ src,
                       const long long* __restrict__ dst,
                       const float* __restrict__ dinv,
                       const float* __restrict__ hlin,
                       float* __restrict__ agg) {
  const int wid = (blockIdx.x * blockDim.x + threadIdx.x) >> 5;
  if (wid >= NE) return;
  const int lane = threadIdx.x & 31;
  const int s = (int)src[wid], d = (int)dst[wid];
  const float nv = dinv[s] * dinv[d];
  const float4 v = *(const float4*)&hlin[(size_t)s * CH + lane * 4];
  float* ap = &agg[(size_t)d * CH + lane * 4];
  unsafeAtomicAdd(ap + 0, v.x * nv);
  unsafeAtomicAdd(ap + 1, v.y * nv);
  unsafeAtomicAdd(ap + 2, v.z * nv);
  unsafeAtomicAdd(ap + 3, v.w * nv);
}

// two channels per thread -> packed bf16 convert
__global__ void k_relu_bf16(const float* __restrict__ agg,
                            const float* __restrict__ b1,
                            unsigned int* __restrict__ hbf2) {
  int p = blockIdx.x * blockDim.x + threadIdx.x;     // dword index
  if (p < NN * CH / 2) {
    const int idx = p * 2;
    const int c = idx & (CH - 1);
    float v0 = fmaxf(agg[idx] + b1[c], 0.0f);
    float v1 = fmaxf(agg[idx + 1] + b1[c + 1], 0.0f);
    hbf2[p] = pack_bf16(v0, v1);
  }
}

// ------------------- GEMM2: loglin = h @ W2  (N=40 -> 3 N-tiles) -----------

__global__ __launch_bounds__(256) void k_gemm2(const unsigned short* __restrict__ hbf,
                                               const float* __restrict__ W2,
                                               float* __restrict__ ll) {
  const int lane = threadIdx.x & 31;
  const int wave = threadIdx.x >> 5;
  const int half = lane >> 4;
  const int mr   = lane & 15;

  Frag16 bfr[3][4];                            // 3 N-tiles x 4 K-chunks
#pragma unroll
  for (int nt = 0; nt < 3; ++nt) {
    const int n = nt * 16 + mr;
#pragma unroll
    for (int c = 0; c < 4; ++c)
#pragma unroll
      for (int j = 0; j < 8; ++j) {
        const int k = 32 * c + 8 * half + 2 * j + (j >= 4 ? 8 : 0);
        bfr[nt][c].u[j] = (n < CL)
            ? pack_bf16(W2[k * CL + n], W2[(k + 1) * CL + n])
            : 0u;
      }
  }

  for (int tile = blockIdx.x * 8 + wave; tile < NN / 16; tile += gridDim.x * 8) {
    const int m0 = tile * 16;
    v8f a0 = {}, a1 = {}, a2 = {};
    Frag16 a;
    {
      const size_t kb = (size_t)(m0 + mr) * CH + 8 * half;
      *(uint4*)&a.u[0] = *(const uint4*)&hbf[kb];        // global_load_b128
      *(uint4*)&a.u[4] = *(const uint4*)&hbf[kb + 16];
    }
#pragma unroll
    for (int c = 0; c < 4; ++c) {
      Frag16 n;                                  // prefetch next K chunk
      if (c < 3) {
        const size_t kb = (size_t)(m0 + mr) * CH + 32 * (c + 1) + 8 * half;
        *(uint4*)&n.u[0] = *(const uint4*)&hbf[kb];
        *(uint4*)&n.u[4] = *(const uint4*)&hbf[kb + 16];
      }
      a0 = __builtin_amdgcn_wmma_f32_16x16x32_bf16(false, a.v, false, bfr[0][c].v,
                                                   (short)0, a0, false, false);
      a1 = __builtin_amdgcn_wmma_f32_16x16x32_bf16(false, a.v, false, bfr[1][c].v,
                                                   (short)0, a1, false, false);
      a2 = __builtin_amdgcn_wmma_f32_16x16x32_bf16(false, a.v, false, bfr[2][c].v,
                                                   (short)0, a2, false, false);
      if (c < 3) a = n;
    }
#pragma unroll
    for (int r = 0; r < 8; ++r) {
      const int row = m0 + r + 8 * half;
      ll[(size_t)row * CL + mr]      = a0[r];            // n = mr      < 40
      ll[(size_t)row * CL + 16 + mr] = a1[r];            // n = 16+mr   < 40
      if (mr < 8) ll[(size_t)row * CL + 32 + mr] = a2[r];
    }
  }
}

// ------------------------- layer-2 aggregation -----------------------------

__global__ void k_selfinit2(const float* __restrict__ ll,
                            const float* __restrict__ dinv,
                            float* __restrict__ agg) {
  int idx = blockIdx.x * blockDim.x + threadIdx.x;
  if (idx < NN * CL) {
    int i = idx / CL;
    float dv = dinv[i];
    agg[idx] = ll[idx] * dv * dv;
  }
}

__global__ void k_agg2(const long long* __restrict__ src,
                       const long long* __restrict__ dst,
                       const float* __restrict__ dinv,
                       const float* __restrict__ ll,
                       float* __restrict__ agg) {
  const int wid = (blockIdx.x * blockDim.x + threadIdx.x) >> 5;
  if (wid >= NE) return;
  const int lane = threadIdx.x & 31;
  const int s = (int)src[wid], d = (int)dst[wid];
  const float nv = dinv[s] * dinv[d];
  unsafeAtomicAdd(&agg[(size_t)d * CL + lane], ll[(size_t)s * CL + lane] * nv);
  if (lane < CL - 32) {
    const int c = 32 + lane;
    unsafeAtomicAdd(&agg[(size_t)d * CL + c], ll[(size_t)s * CL + c] * nv);
  }
}

__global__ void k_logsoftmax(const float* __restrict__ agg,
                             const float* __restrict__ b2,
                             float* __restrict__ out) {
  int i = blockIdx.x * blockDim.x + threadIdx.x;
  if (i >= NN) return;
  float v[CL];
  float m = -3.4e38f;
#pragma unroll
  for (int c = 0; c < CL; ++c) {
    v[c] = agg[(size_t)i * CL + c] + b2[c];
    m = fmaxf(m, v[c]);
  }
  float s = 0.0f;
#pragma unroll
  for (int c = 0; c < CL; ++c) s += __expf(v[c] - m);
  const float ls = __logf(s);
#pragma unroll
  for (int c = 0; c < CL; ++c) out[(size_t)i * CL + c] = v[c] - m - ls;
}

// ------------------------------- driver ------------------------------------

extern "C" void kernel_launch(void* const* d_in, const int* in_sizes, int n_in,
                              void* d_out, int out_size, void* d_ws, size_t ws_size,
                              hipStream_t stream) {
  const float*     x  = (const float*)d_in[0];
  const long long* ei = (const long long*)d_in[1];   // int64 [2, NE]
  const float*     W1 = (const float*)d_in[2];
  const float*     b1 = (const float*)d_in[3];
  const float*     W2 = (const float*)d_in[4];
  const float*     b2 = (const float*)d_in[5];
  float*           out = (float*)d_out;

  const long long* esrc = ei;
  const long long* edst = ei + NE;

  // workspace carve-out (~206 MB with lifetime-based aliasing)
  char* ws = (char*)d_ws;
  size_t off = 0;
  auto carve = [&](size_t bytes) {
    void* p = ws + off;
    off = (off + bytes + 255) & ~(size_t)255;
    return p;
  };
  float*          deg  = (float*)carve((size_t)NN * 4);
  float*          dinv = (float*)carve((size_t)NN * 4);
  unsigned short* xbf  = (unsigned short*)carve((size_t)NN * CIN * 2);  // 102 MB
  float*          hlin = (float*)carve((size_t)NN * CH * 4);            // 51 MB
  float*          agg1 = (float*)carve((size_t)NN * CH * 4);            // 51 MB
  // aliases: xbf is dead after k_gemm1, hlin is dead after k_agg1
  float*          ll   = (float*)xbf;                                   // 16 MB
  float*          agg2 = (float*)((char*)xbf + ((size_t)32 << 20));     // 16 MB
  unsigned short* hbf  = (unsigned short*)hlin;                         // 25 MB

  const int T = 256;
  k_deg_init  <<<(NN + T - 1) / T, T, 0, stream>>>(deg);
  k_deg_edges <<<(NE + T - 1) / T, T, 0, stream>>>(edst, deg);
  k_dinv      <<<(NN + T - 1) / T, T, 0, stream>>>(deg, dinv);

  k_cvt_x     <<<NN * CIN / 8 / T, T, 0, stream>>>(x, (uint4*)xbf);
  k_gemm1     <<<1600, T, 0, stream>>>(xbf, W1, hlin);
  k_selfinit1 <<<(NN * CH + T - 1) / T, T, 0, stream>>>(hlin, dinv, agg1);
  k_agg1      <<<NE * 32 / T, T, 0, stream>>>(esrc, edst, dinv, hlin, agg1);
  k_relu_bf16 <<<(NN * CH / 2 + T - 1) / T, T, 0, stream>>>(agg1, b1,
                                                            (unsigned int*)hbf);

  k_gemm2     <<<782, T, 0, stream>>>(hbf, W2, ll);
  k_selfinit2 <<<(NN * CL + T - 1) / T, T, 0, stream>>>(ll, dinv, agg2);
  k_agg2      <<<NE * 32 / T, T, 0, stream>>>(esrc, edst, dinv, ll, agg2);
  k_logsoftmax<<<(NN + T - 1) / T, T, 0, stream>>>(agg2, b2, out);
}
